// AttentionAggregator_41412074668237
// MI455X (gfx1250) — compile-verified, matrix operand order checked by reading, not run on hardware
//
#include <hip/hip_runtime.h>
#include <hip/hip_bf16.h>
#include <climits>

#define N_NODES 100000
#define N_EDGES 1600000
#define IN_DIM  256
#define OUT_DIM 128
#define BATCH   8192
#define SLOPE   0.1f
#define EPS_    1e-8f

typedef __attribute__((ext_vector_type(16))) __bf16 v16bf;
typedef __attribute__((ext_vector_type(8)))  float  v8f;

__device__ __forceinline__ void bsplit(float x, __bf16& hi, __bf16& lo) {
  hi = (__bf16)x;                    // RNE to bf16
  lo = (__bf16)(x - (float)hi);      // exact residual, then bf16
}

// ---------------------------------------------------------------------------
// 0. init workspace (slot map, accumulators, counter)
// ---------------------------------------------------------------------------
__global__ void gat_init(int* __restrict__ slot, float* __restrict__ acc,
                         float* __restrict__ denom, int* __restrict__ count) {
  int i = blockIdx.x * blockDim.x + threadIdx.x;
  int stride = gridDim.x * blockDim.x;
  for (int j = i; j < N_NODES; j += stride) slot[j] = INT_MAX;
  for (int j = i; j < BATCH * OUT_DIM; j += stride) acc[j] = 0.f;
  for (int j = i; j < BATCH; j += stride) denom[j] = 0.f;
  if (i == 0) *count = 0;
}

// ---------------------------------------------------------------------------
// 1. slot[node] = min batch position that references it (deterministic)
// ---------------------------------------------------------------------------
__global__ void gat_slot(const int* __restrict__ batch_idx, int* __restrict__ slot) {
  int i = blockIdx.x * blockDim.x + threadIdx.x;
  if (i < BATCH) atomicMin(&slot[batch_idx[i]], i);
}

// ---------------------------------------------------------------------------
// 2. new_emb = feats @ W + b  (split-bf16 WMMA, fp32 accumulate)
//    fused: s_src = new_emb @ a[:128], s_dst = new_emb @ a[128:]
//    One wave computes a 16x128 strip; 8 waves / block; W swizzled in LDS.
// ---------------------------------------------------------------------------
__global__ void __launch_bounds__(256)
gat_gemm(const float* __restrict__ feats, const float* __restrict__ W,
         const float* __restrict__ b, const float* __restrict__ a,
         float* __restrict__ emb, float* __restrict__ s_src,
         float* __restrict__ s_dst) {
  extern __shared__ __align__(32) char smem_raw[];
  __bf16* Wh = (__bf16*)smem_raw;            // 256*128 bf16 hi  (64 KB)
  __bf16* Wl = Wh + IN_DIM * OUT_DIM;        // 256*128 bf16 lo  (64 KB)

  const int tid = threadIdx.x;
  // stage W, split hi/lo, swizzle into B-fragment order:
  // frag element i of lane l for (kstep ks, ntile nt):
  //   k = ks*32 + (i<8?0:16) + (l>=16?8:0) + (i&7),  n = nt*16 + (l&15)
  for (int idx = tid; idx < IN_DIM * OUT_DIM; idx += 256) {
    int k = idx >> 7;
    int n = idx & 127;
    __bf16 hi, lo;
    bsplit(W[idx], hi, lo);
    int ks = k >> 5, kr = k & 31;
    int hh = (kr >> 3) & 1;
    int i  = (kr & 7) | ((kr >> 4) << 3);
    int ln = (n & 15) | (hh << 4);
    int nt = n >> 4;
    int off = (((ks * 8 + nt) * 32 + ln) << 4) + i;
    Wh[off] = hi;
    Wl[off] = lo;
  }
  __syncthreads();

  const int lane  = tid & 31;
  const int wv    = tid >> 5;
  const int strip = blockIdx.x * 8 + wv;
  if (strip >= N_NODES / 16) return;

  const int hh    = lane >> 4;
  const int rbase = strip * 16;
  const float* arow = feats + (size_t)(rbase + (lane & 15)) * IN_DIM;

  v8f acc[8] = {};

  #pragma unroll
  for (int ks = 0; ks < 8; ++ks) {
    const int k0 = ks * 32 + hh * 8;
    float av[16];
    *(float4*)(&av[0])  = *(const float4*)(arow + k0);
    *(float4*)(&av[4])  = *(const float4*)(arow + k0 + 4);
    *(float4*)(&av[8])  = *(const float4*)(arow + k0 + 16);
    *(float4*)(&av[12]) = *(const float4*)(arow + k0 + 20);
    v16bf ahi, alo;
    #pragma unroll
    for (int i = 0; i < 16; ++i) {
      __bf16 h, l;
      bsplit(av[i], h, l);
      ahi[i] = h;
      alo[i] = l;
    }
    #pragma unroll
    for (int nt = 0; nt < 8; ++nt) {
      const int boff = (((ks * 8 + nt) * 32 + lane) << 4);
      v16bf bhi = *(const v16bf*)(Wh + boff);
      v16bf blo = *(const v16bf*)(Wl + boff);
      acc[nt] = __builtin_amdgcn_wmma_f32_16x16x32_bf16(
          false, ahi, false, bhi, (short)0, acc[nt], false, false);
      acc[nt] = __builtin_amdgcn_wmma_f32_16x16x32_bf16(
          false, ahi, false, blo, (short)0, acc[nt], false, false);
      acc[nt] = __builtin_amdgcn_wmma_f32_16x16x32_bf16(
          false, alo, false, bhi, (short)0, acc[nt], false, false);
    }
  }

  // epilogue: +bias, store emb, fused score dots (C/D layout: row = v + hh*8)
  float psrc[8], pdst[8];
  #pragma unroll
  for (int v = 0; v < 8; ++v) { psrc[v] = 0.f; pdst[v] = 0.f; }
  #pragma unroll
  for (int nt = 0; nt < 8; ++nt) {
    int n = nt * 16 + (lane & 15);
    float bias = b[n];
    float a0 = a[n];
    float a1 = a[OUT_DIM + n];
    #pragma unroll
    for (int v = 0; v < 8; ++v) {
      float val = acc[nt][v] + bias;
      int row = rbase + v + hh * 8;
      emb[(size_t)row * OUT_DIM + n] = val;
      psrc[v] = fmaf(val, a0, psrc[v]);
      pdst[v] = fmaf(val, a1, pdst[v]);
    }
  }
  // reduce over the 16 column-lanes (xor 1,2,4,8 stays inside each half)
  #pragma unroll
  for (int m = 1; m <= 8; m <<= 1) {
    #pragma unroll
    for (int v = 0; v < 8; ++v) {
      psrc[v] += __shfl_xor(psrc[v], m, 32);
      pdst[v] += __shfl_xor(pdst[v], m, 32);
    }
  }
  if ((lane & 15) == 0) {
    #pragma unroll
    for (int v = 0; v < 8; ++v) {
      int row = rbase + v + hh * 8;
      s_src[row] = psrc[v];
      s_dst[row] = pdst[v];
    }
  }
}

// ---------------------------------------------------------------------------
// 3. compact edges whose destination row r is in the batch (~12x cut)
// ---------------------------------------------------------------------------
__global__ void gat_filter(const int* __restrict__ r, const int* __restrict__ slot,
                           int* __restrict__ active, int* __restrict__ count) {
  int e = blockIdx.x * blockDim.x + threadIdx.x;
  if (e >= N_EDGES) return;
  if (slot[r[e]] != INT_MAX) {
    int pos = atomicAdd(count, 1);
    active[pos] = e;
  }
}

// ---------------------------------------------------------------------------
// 4. one wave per active edge: w = exp(lrelu(s)), scatter-add w*emb[c] + w
// ---------------------------------------------------------------------------
__global__ void gat_edge(const int* __restrict__ active, const int* __restrict__ count,
                         const int* __restrict__ r, const int* __restrict__ c,
                         const float* __restrict__ s_src, const float* __restrict__ s_dst,
                         const float* __restrict__ emb, const int* __restrict__ slot,
                         float* __restrict__ acc, float* __restrict__ denom) {
  const int lane   = threadIdx.x & 31;
  const int waveId = (blockIdx.x * blockDim.x + threadIdx.x) >> 5;
  const int nWaves = (gridDim.x * blockDim.x) >> 5;
  const int cnt = *count;
  for (int idx = waveId; idx < cnt; idx += nWaves) {
    int e  = active[idx];
    int rr = r[e];
    int cc = c[e];
    float s = s_src[rr] + s_dst[cc];
    float w = expf(s > 0.f ? s : s * SLOPE);
    int sl = slot[rr];
    if (lane == 0) unsafeAtomicAdd(&denom[sl], w);
    const float4 v = *(const float4*)(emb + (size_t)cc * OUT_DIM + lane * 4);
    float* dst = acc + (size_t)sl * OUT_DIM + lane * 4;
    unsafeAtomicAdd(dst + 0, w * v.x);
    unsafeAtomicAdd(dst + 1, w * v.y);
    unsafeAtomicAdd(dst + 2, w * v.z);
    unsafeAtomicAdd(dst + 3, w * v.w);
  }
}

// ---------------------------------------------------------------------------
// 5. out[i] = acc[slot[batch_idx[i]]] / (denom + eps)
// ---------------------------------------------------------------------------
__global__ void gat_final(const int* __restrict__ batch_idx, const int* __restrict__ slot,
                          const float* __restrict__ acc, const float* __restrict__ denom,
                          float* __restrict__ out) {
  int i = blockIdx.x * blockDim.x + threadIdx.x;
  if (i >= BATCH * OUT_DIM) return;
  int row = i >> 7;
  int n   = i & 127;
  int sl  = slot[batch_idx[row]];
  out[i] = acc[(size_t)sl * OUT_DIM + n] / (denom[sl] + EPS_);
}

// ---------------------------------------------------------------------------
extern "C" void kernel_launch(void* const* d_in, const int* in_sizes, int n_in,
                              void* d_out, int out_size, void* d_ws, size_t ws_size,
                              hipStream_t stream) {
  const float* feats     = (const float*)d_in[0];
  const float* W         = (const float*)d_in[1];
  const float* b         = (const float*)d_in[2];
  const float* a         = (const float*)d_in[3];
  const int*   r         = (const int*)d_in[4];
  const int*   c         = (const int*)d_in[5];
  const int*   batch_idx = (const int*)d_in[6];
  float* out = (float*)d_out;

  char* ws = (char*)d_ws;
  size_t off = 0;
  auto carve = [&](size_t bytes) -> void* {
    void* p = ws + off;
    off = (off + bytes + 255) & ~(size_t)255;
    return p;
  };
  float* emb    = (float*)carve((size_t)N_NODES * OUT_DIM * 4);  // 51.2 MB
  float* s_src  = (float*)carve((size_t)N_NODES * 4);
  float* s_dst  = (float*)carve((size_t)N_NODES * 4);
  int*   slot   = (int*)carve((size_t)N_NODES * 4);
  float* acc    = (float*)carve((size_t)BATCH * OUT_DIM * 4);    // 4 MB
  float* denom  = (float*)carve((size_t)BATCH * 4);
  int*   count  = (int*)carve(256);
  int*   active = (int*)carve((size_t)N_EDGES * 4);              // 6.4 MB

  gat_init<<<1024, 256, 0, stream>>>(slot, acc, denom, count);
  gat_slot<<<BATCH / 256, 256, 0, stream>>>(batch_idx, slot);
  // 6250 strips of 16 rows, 8 strips per block
  gat_gemm<<<(N_NODES / 16 + 7) / 8, 256, 2 * IN_DIM * OUT_DIM * sizeof(__bf16), stream>>>(
      feats, W, b, a, emb, s_src, s_dst);
  gat_filter<<<N_EDGES / 256, 256, 0, stream>>>(r, slot, active, count);
  gat_edge<<<2048, 256, 0, stream>>>(active, count, r, c, s_src, s_dst, emb, slot, acc, denom);
  gat_final<<<(BATCH * OUT_DIM) / 256, 256, 0, stream>>>(batch_idx, slot, acc, denom, out);
}
